// WorkerActNet_55250459296473
// MI455X (gfx1250) — matrix operand order repaired and checked by reading, not compiled, not run on hardware
//
#include <hip/hip_runtime.h>
#include <math.h>

// ---------------------------------------------------------------------------
// Model dims
// ---------------------------------------------------------------------------
#define BB   256          // batch
#define TT   512          // time
#define CC   27           // channels
#define KKW  5            // conv kernel
#define HH   1024         // hidden
#define T2   256          // time after pool
#define NG   4096         // 4*H gate width
#define NT_LSTM 256       // NG/16  n-tiles in packed LSTM weights
#define KT_LSTM 33        // 32 k-tiles for W_hh + 1 k-tile for padded W_ih

typedef __attribute__((ext_vector_type(16))) __bf16 v16bf;
typedef __attribute__((ext_vector_type(8)))  float  v8f;
typedef __attribute__((ext_vector_type(4)))  unsigned int u32x4;
typedef __attribute__((ext_vector_type(8)))  int i32x8;
typedef __attribute__((ext_vector_type(4)))  int i32x4;

__device__ inline unsigned short f2bf(float f) {
  unsigned int u = __float_as_uint(f);
  unsigned int r = (u + 0x7FFFu + ((u >> 16) & 1u)) >> 16;
  return (unsigned short)r;
}

union FragBF { v16bf v; uint4 q[2]; };

// A fragment: 16x32 bf16, row-major source with leading dim lda (bf16 elems).
__device__ inline FragBF load_frag_A(const unsigned short* __restrict__ A, int lda,
                                     int m0, int k0, int lane) {
  FragBF f;
  int r  = lane & 15;
  int c0 = (lane < 16) ? 0 : 16;
  const unsigned short* p = A + (size_t)(m0 + r) * lda + k0 + c0;
  f.q[0] = *reinterpret_cast<const uint4*>(p);
  f.q[1] = *reinterpret_cast<const uint4*>(p + 8);
  return f;
}

// B fragment from pre-swizzled tiles in global memory (32 lanes x 32B per tile).
__device__ inline FragBF load_frag_B(const unsigned short* __restrict__ Bp,
                                     int tile, int lane) {
  FragBF f;
  const unsigned short* p = Bp + ((size_t)tile * 32 + lane) * 16;
  f.q[0] = *reinterpret_cast<const uint4*>(p);
  f.q[1] = *reinterpret_cast<const uint4*>(p + 8);
  return f;
}

// B fragment from an LDS staging buffer (16 contiguous bf16 per lane).
__device__ inline FragBF load_frag_B_lds(const unsigned short* p) {
  FragBF f;
  f.q[0] = *reinterpret_cast<const uint4*>(p);
  f.q[1] = *reinterpret_cast<const uint4*>(p + 8);
  return f;
}

__device__ inline v8f wmma_bf16(FragBF a, FragBF b, v8f c) {
  return __builtin_amdgcn_wmma_f32_16x16x32_bf16(false, a.v, false, b.v,
                                                 (short)0, c, false, false);
}

// ---------------------------------------------------------------------------
// TDM: one 2D descriptor pulls the block's 8 packed B tiles (4 gate rows of
// 2KB, row stride 64KB) for one K-step into LDS.  data_size = 8B units.
// ---------------------------------------------------------------------------
__device__ inline void tdm_load_stage(const unsigned short* __restrict__ Bp,
                                      int kt, int ntb2, unsigned lds_off) {
  unsigned long long gaddr =
      (unsigned long long)(uintptr_t)Bp + ((size_t)kt * NT_LSTM + ntb2) * 1024ull;
  u32x4 g0;
  g0.x = 1u;                                        // count=1, user descriptor
  g0.y = lds_off;                                   // LDS byte address
  g0.z = (unsigned)(gaddr & 0xffffffffu);           // global_addr[31:0]
  g0.w = (unsigned)((gaddr >> 32) & 0x1ffffffu) | (2u << 30);  // addr[56:32]|type=2
  i32x8 g1;
  g1[0] = 0x30000;        // workgroup_mask=0, data_size=3 (8B)
  g1[1] = 0x20000000;     // tensor_dim0 = 8192 units (low16 << 16)
  g1[2] = 0x400000;       // tensor_dim1 = 64 (low16 << 16)
  g1[3] = 0x1000000;      // tile_dim0 = 256 units (2KB row)
  g1[4] = 4;              // tile_dim1 = 4 rows (the 4 gates)
  g1[5] = 0x2000;         // tensor_dim0_stride = 8192 units (64KB)
  g1[6] = 0;
  g1[7] = 0;
  i32x4 gz4 = {0, 0, 0, 0};
  i32x8 gz8 = {0, 0, 0, 0, 0, 0, 0, 0};
  __builtin_amdgcn_tensor_load_to_lds(g0, g1, gz4, gz4, gz8, 0);
}

// ---------------------------------------------------------------------------
// Utility kernels
// ---------------------------------------------------------------------------
__global__ void k_zero_f32(float* p, int n) {
  int i = blockIdx.x * blockDim.x + threadIdx.x;
  if (i < n) p[i] = 0.f;
}
__global__ void k_zero_u16(unsigned short* p, int n) {
  int i = blockIdx.x * blockDim.x + threadIdx.x;
  if (i < n) p[i] = 0;
}
__global__ void k_bias4(const float* __restrict__ bih, const float* __restrict__ bhh,
                        float* __restrict__ bias4) {
  int i = blockIdx.x * blockDim.x + threadIdx.x;
  if (i < NG) bias4[i] = bih[i] + bhh[i];
}

// ---------------------------------------------------------------------------
// Depthwise conv (pad 2) -> y stored transposed [C][B*T] for deterministic stats
// ---------------------------------------------------------------------------
__global__ void k_conv(const float* __restrict__ x, const float* __restrict__ w,
                       const float* __restrict__ cb, float* __restrict__ yws) {
  int idx = blockIdx.x * blockDim.x + threadIdx.x;   // b*T + t
  int b = idx >> 9;
  int t = idx & 511;
  float acc[CC];
#pragma unroll
  for (int c = 0; c < CC; ++c) acc[c] = cb[c];
#pragma unroll
  for (int k = 0; k < KKW; ++k) {
    int tk = t + k - 2;
    if (tk >= 0 && tk < TT) {
      const float* xr = x + ((size_t)b * TT + tk) * CC;
      for (int c = 0; c < CC; ++c) acc[c] += w[c * KKW + k] * xr[c];
    }
  }
  for (int c = 0; c < CC; ++c) yws[(size_t)c * (BB * TT) + idx] = acc[c];
}

// Per-channel mean/var (deterministic tree reduction) -> scale/shift
__global__ void k_stats(const float* __restrict__ yws, const float* __restrict__ gamma,
                        const float* __restrict__ beta, float* __restrict__ scale,
                        float* __restrict__ shiftv) {
  __shared__ float ssum[256];
  __shared__ float ssq[256];
  int c = blockIdx.x;
  const float* y = yws + (size_t)c * (BB * TT);
  float s = 0.f, q = 0.f;
  for (int i = threadIdx.x; i < BB * TT; i += 256) {
    float v = y[i];
    s += v; q += v * v;
  }
  ssum[threadIdx.x] = s; ssq[threadIdx.x] = q;
  __syncthreads();
  for (int off = 128; off > 0; off >>= 1) {
    if (threadIdx.x < off) {
      ssum[threadIdx.x] += ssum[threadIdx.x + off];
      ssq[threadIdx.x]  += ssq[threadIdx.x + off];
    }
    __syncthreads();
  }
  if (threadIdx.x == 0) {
    float n = (float)(BB * TT);
    float mean = ssum[0] / n;
    float var  = ssq[0] / n - mean * mean;
    float sc = gamma[c] * rsqrtf(var + 1e-5f);
    scale[c]  = sc;
    shiftv[c] = beta[c] - mean * sc;
  }
}

// BN + ELU + MaxPool(2) -> bf16 xpad[t2][b][32] (channels padded 27->32)
__global__ void k_apply(const float* __restrict__ yws, const float* __restrict__ scale,
                        const float* __restrict__ shiftv,
                        unsigned short* __restrict__ xpad) {
  int idx = blockIdx.x * blockDim.x + threadIdx.x;   // b*T2 + t2
  int b = idx >> 8;
  int t2 = idx & 255;
  unsigned short* dst = xpad + ((size_t)t2 * BB + b) * 32;
  for (int c = 0; c < CC; ++c) {
    size_t base = (size_t)c * (BB * TT) + (size_t)b * TT + 2 * t2;
    float y0 = yws[base];
    float y1 = yws[base + 1];
    y0 = scale[c] * y0 + shiftv[c];
    y1 = scale[c] * y1 + shiftv[c];
    y0 = y0 > 0.f ? y0 : expm1f(y0);
    y1 = y1 > 0.f ? y1 : expm1f(y1);
    dst[c] = f2bf(fmaxf(y0, y1));
  }
  for (int c = CC; c < 32; ++c) dst[c] = 0;
}

// ---------------------------------------------------------------------------
// Weight packing into WMMA B-fragment layout
// tile(kt,nt): lane L holds 16 contiguous bf16 = B[kt*32 + (L<16?0:16) + e][nt*16 + (L&15)]
// where B[k][n] = W[n][k] (we compute h @ W^T).
// ---------------------------------------------------------------------------
__global__ void k_pack_lstm(const float* __restrict__ whh, const float* __restrict__ wih,
                            unsigned short* __restrict__ Bp) {
  int gid = blockIdx.x * blockDim.x + threadIdx.x;   // tile*32 + lane
  if (gid >= KT_LSTM * NT_LSTM * 32) return;
  int lane = gid & 31;
  int tile = gid >> 5;
  int nt = tile % NT_LSTM;
  int kt = tile / NT_LSTM;
  int n = nt * 16 + (lane & 15);
  int kb = (lane < 16) ? 0 : 16;
  unsigned short* dst = Bp + (size_t)gid * 16;
#pragma unroll
  for (int e = 0; e < 16; ++e) {
    int kl = kb + e;
    float v;
    if (kt < 32) v = whh[(size_t)n * HH + kt * 32 + kl];
    else         v = (kl < CC) ? wih[(size_t)n * CC + kl] : 0.f;   // padded x tile
    dst[e] = f2bf(v);
  }
}

__global__ void k_pack_b(const float* __restrict__ W, unsigned short* __restrict__ Bp,
                         int N, int K) {
  int gid = blockIdx.x * blockDim.x + threadIdx.x;
  int ktiles = K >> 5, ntiles = N >> 4;
  if (gid >= ktiles * ntiles * 32) return;
  int lane = gid & 31;
  int tile = gid >> 5;
  int nt = tile % ntiles;
  int kt = tile / ntiles;
  int n = nt * 16 + (lane & 15);
  int kb = kt * 32 + ((lane < 16) ? 0 : 16);
  unsigned short* dst = Bp + (size_t)gid * 16;
#pragma unroll
  for (int e = 0; e < 16; ++e) dst[e] = f2bf(W[(size_t)n * K + kb + e]);
}

// ---------------------------------------------------------------------------
// One LSTM time step.  Block = 8 waves, tile 64(M) x 32(N_h), grid 128.
// B operands stream through a double-buffered LDS stage filled by the TDM
// (one tensor_load_to_lds per K-step, overlapped with the current WMMAs);
// A fragments are software-pipelined in registers.
// ---------------------------------------------------------------------------
__global__ void __launch_bounds__(256)
k_lstm_step(const unsigned short* __restrict__ h_in,   // [256][1024] bf16
            const unsigned short* __restrict__ x_t,    // [256][32]   bf16
            const unsigned short* __restrict__ Bp,     // packed [33][256] tiles
            const float* __restrict__ bias,            // [4096]
            float* __restrict__ cst,                   // [256][1024] fp32
            unsigned short* __restrict__ h_out) {      // [256][1024] bf16
  __shared__ alignas(16) unsigned short sB[2][4096];   // 2 x 8KB B staging
  int lane = threadIdx.x & 31;
  int w    = threadIdx.x >> 5;
  int bm = blockIdx.x & 3;
  int bn = blockIdx.x >> 2;
  int m0  = bm * 64 + (w & 3) * 16;
  int nh0 = bn * 32 + (w >> 2) * 16;
  int ntb2  = bn * 2;       // first of the two 16-wide n-tiles of this block
  int ntoff = w >> 2;       // 0/1: which n-tile this wave consumes

  unsigned lds0 = (unsigned)(uintptr_t)(&sB[0][0]);
  unsigned lds1 = (unsigned)(uintptr_t)(&sB[1][0]);

  v8f a0 = {}, a1 = {}, a2 = {}, a3 = {};

  if (w == 0) tdm_load_stage(Bp, 0, ntb2, lds0);
  FragBF a = load_frag_A(h_in, HH, m0, 0, lane);
  if (w == 0) __builtin_amdgcn_s_wait_tensorcnt(0);
  __syncthreads();

  for (int kt = 0; kt < KT_LSTM; ++kt) {
    int cur = kt & 1;
    if (w == 0 && kt + 1 < KT_LSTM)
      tdm_load_stage(Bp, kt + 1, ntb2, cur ? lds0 : lds1);
    FragBF an = a;
    if (kt + 1 < 32)        an = load_frag_A(h_in, HH, m0, (kt + 1) * 32, lane);
    else if (kt + 1 == 32)  an = load_frag_A(x_t, 32, m0, 0, lane);
    const unsigned short* sb = &sB[cur][ntoff * 512 + lane * 16];
    FragBF b0 = load_frag_B_lds(sb);            // gate i  (row 0, +0KB)
    FragBF b1 = load_frag_B_lds(sb + 1024);     // gate f  (row 1, +2KB)
    FragBF b2 = load_frag_B_lds(sb + 2048);     // gate g  (row 2, +4KB)
    FragBF b3 = load_frag_B_lds(sb + 3072);     // gate o  (row 3, +6KB)
    a0 = wmma_bf16(a, b0, a0);
    a1 = wmma_bf16(a, b1, a1);
    a2 = wmma_bf16(a, b2, a2);
    a3 = wmma_bf16(a, b3, a3);
    a = an;
    if (kt + 1 < KT_LSTM) {
      if (w == 0) __builtin_amdgcn_s_wait_tensorcnt(0);
      __syncthreads();
    }
  }

  int j = nh0 + (lane & 15);
  int mofs = (lane < 16) ? 0 : 8;
  float bi = bias[j];
  float bf = bias[HH + j];
  float bg = bias[2 * HH + j];
  float bo = bias[3 * HH + j];
#pragma unroll
  for (int r = 0; r < 8; ++r) {
    int m = m0 + mofs + r;
    float ig = 1.f / (1.f + __expf(-(a0[r] + bi)));
    float fg = 1.f / (1.f + __expf(-(a1[r] + bf)));
    float gg = tanhf(a2[r] + bg);
    float og = 1.f / (1.f + __expf(-(a3[r] + bo)));
    size_t o = (size_t)m * HH + j;
    float cn = fg * cst[o] + ig * gg;
    cst[o] = cn;
    h_out[o] = f2bf(og * tanhf(cn));
  }
}

// ---------------------------------------------------------------------------
// Generic WMMA GEMM (heads): out = A[256,K](bf16) @ Wpack + bias
// Software-pipelined register path.
// ---------------------------------------------------------------------------
__global__ void __launch_bounds__(256)
k_gemm_bias(const unsigned short* __restrict__ A,
            const unsigned short* __restrict__ Bp,
            const float* __restrict__ bias,
            unsigned short* __restrict__ outb,
            float* __restrict__ outf,
            int N, int K) {
  int lane = threadIdx.x & 31;
  int w    = threadIdx.x >> 5;
  int bm = blockIdx.x & 3;
  int bn = blockIdx.x >> 2;
  int m0 = bm * 64 + (w & 3) * 16;
  int n0 = bn * 32 + (w >> 2) * 16;
  int ntiles = N >> 4;
  int nt = n0 >> 4;
  int ktiles = K >> 5;
  v8f acc = {};
  FragBF a = load_frag_A(A, K, m0, 0, lane);
  FragBF b = load_frag_B(Bp, nt, lane);
  for (int kt = 0; kt + 1 < ktiles; ++kt) {
    FragBF an = load_frag_A(A, K, m0, (kt + 1) * 32, lane);
    FragBF bn = load_frag_B(Bp, (kt + 1) * ntiles + nt, lane);
    acc = wmma_bf16(a, b, acc);
    a = an; b = bn;
  }
  acc = wmma_bf16(a, b, acc);
  int j = n0 + (lane & 15);
  int mofs = (lane < 16) ? 0 : 8;
  float bj = bias[j];
#pragma unroll
  for (int r = 0; r < 8; ++r) {
    int m = m0 + mofs + r;
    float v = acc[r] + bj;
    if (outb) outb[(size_t)m * N + j] = f2bf(v);
    if (outf) outf[(size_t)m * N + j] = v;
  }
}

// Final narrow layers in fp32: cls[256,16] then p1[256,1] into d_out.
__global__ void k_head_final(const float* __restrict__ vc, const float* __restrict__ wc3,
                             const float* __restrict__ bc3,
                             const float* __restrict__ vp, const float* __restrict__ wp3,
                             const float* __restrict__ bp3,
                             float* __restrict__ out) {
  int lane = threadIdx.x & 31;
  int wid = (blockIdx.x * blockDim.x + threadIdx.x) >> 5;
  if (wid >= BB * 17) return;
  int row = wid / 17;
  int o   = wid % 17;
  const float* a;
  const float* wv;
  float bsc;
  if (o < 16) { a = vc + (size_t)row * HH; wv = wc3 + (size_t)o * HH; bsc = bc3[o]; }
  else        { a = vp + (size_t)row * HH; wv = wp3;                  bsc = bp3[0]; }
  float s = 0.f;
  for (int i = lane; i < HH; i += 32) s += a[i] * wv[i];
  for (int off = 16; off > 0; off >>= 1) s += __shfl_xor(s, off, 32);
  if (lane == 0) {
    if (o < 16) out[row * 16 + o] = s + bsc;
    else        out[NG + row]     = s + bsc;
  }
}

// ---------------------------------------------------------------------------
// Host launch
// ---------------------------------------------------------------------------
extern "C" void kernel_launch(void* const* d_in, const int* in_sizes, int n_in,
                              void* d_out, int out_size, void* d_ws, size_t ws_size,
                              hipStream_t stream) {
  (void)in_sizes; (void)n_in; (void)out_size; (void)ws_size;
  const float* x      = (const float*)d_in[0];
  const float* conv_w = (const float*)d_in[1];
  const float* conv_b = (const float*)d_in[2];
  const float* gamma  = (const float*)d_in[3];
  const float* beta   = (const float*)d_in[4];
  const float* w_ih   = (const float*)d_in[5];
  const float* w_hh   = (const float*)d_in[6];
  const float* b_ih   = (const float*)d_in[7];
  const float* b_hh   = (const float*)d_in[8];
  const float* wc1 = (const float*)d_in[9];   const float* bc1 = (const float*)d_in[10];
  const float* wc2 = (const float*)d_in[11];  const float* bc2 = (const float*)d_in[12];
  const float* wc3 = (const float*)d_in[13];  const float* bc3 = (const float*)d_in[14];
  const float* wp1 = (const float*)d_in[15];  const float* bp1 = (const float*)d_in[16];
  const float* wp2 = (const float*)d_in[17];  const float* bp2 = (const float*)d_in[18];
  const float* wp3 = (const float*)d_in[19];  const float* bp3 = (const float*)d_in[20];
  float* out = (float*)d_out;

  char* ws = (char*)d_ws;
  size_t off = 0;
  auto alloc = [&](size_t bytes) -> void* {
    void* p = ws + off;
    off += (bytes + 255) & ~(size_t)255;
    return p;
  };
  float* yws   = (float*)alloc(sizeof(float) * (size_t)CC * BB * TT);
  float* scale = (float*)alloc(sizeof(float) * 32);
  float* shv   = (float*)alloc(sizeof(float) * 32);
  float* bias4 = (float*)alloc(sizeof(float) * NG);
  unsigned short* BpL  = (unsigned short*)alloc(2ull * KT_LSTM * NT_LSTM * 32 * 16);
  unsigned short* Bpc1 = (unsigned short*)alloc(2ull * 32 * 64 * 32 * 16);
  unsigned short* Bpc2 = (unsigned short*)alloc(2ull * 32 * 64 * 32 * 16);
  unsigned short* Bpp1 = (unsigned short*)alloc(2ull * 32 * 64 * 32 * 16);
  unsigned short* Bpp2 = (unsigned short*)alloc(2ull * 32 * 64 * 32 * 16);
  unsigned short* xpad = (unsigned short*)alloc(2ull * T2 * BB * 32);
  unsigned short* h0   = (unsigned short*)alloc(2ull * BB * HH);
  unsigned short* h1   = (unsigned short*)alloc(2ull * BB * HH);
  float* cbuf = (float*)alloc(sizeof(float) * BB * HH);
  unsigned short* u_c = (unsigned short*)alloc(2ull * BB * HH);
  unsigned short* u_p = (unsigned short*)alloc(2ull * BB * HH);
  float* v_c = (float*)alloc(sizeof(float) * BB * HH);
  float* v_p = (float*)alloc(sizeof(float) * BB * HH);

  // State init (re-done every call -> deterministic)
  k_zero_f32<<<(BB * HH + 255) / 256, 256, 0, stream>>>(cbuf, BB * HH);
  k_zero_u16<<<(BB * HH + 255) / 256, 256, 0, stream>>>(h0, BB * HH);
  k_bias4<<<NG / 256, 256, 0, stream>>>(b_ih, b_hh, bias4);

  // Frontend: conv -> stats -> BN/ELU/pool -> bf16 xpad
  k_conv<<<(BB * TT) / 256, 256, 0, stream>>>(x, conv_w, conv_b, yws);
  k_stats<<<CC, 256, 0, stream>>>(yws, gamma, beta, scale, shv);
  k_apply<<<(BB * T2) / 256, 256, 0, stream>>>(yws, scale, shv, xpad);

  // Weight packing (bf16, WMMA B-fragment layout)
  k_pack_lstm<<<(KT_LSTM * NT_LSTM * 32) / 256, 256, 0, stream>>>(w_hh, w_ih, BpL);
  k_pack_b<<<(32 * 64 * 32) / 256, 256, 0, stream>>>(wc1, Bpc1, 1024, 1024);
  k_pack_b<<<(32 * 64 * 32) / 256, 256, 0, stream>>>(wc2, Bpc2, 1024, 1024);
  k_pack_b<<<(32 * 64 * 32) / 256, 256, 0, stream>>>(wp1, Bpp1, 1024, 1024);
  k_pack_b<<<(32 * 64 * 32) / 256, 256, 0, stream>>>(wp2, Bpp2, 1024, 1024);

  // LSTM recurrence: 256 graph-captured step launches, double-buffered h
  unsigned short* hb[2] = {h0, h1};
  for (int t = 0; t < T2; ++t) {
    k_lstm_step<<<128, 256, 0, stream>>>(hb[t & 1], xpad + (size_t)t * BB * 32,
                                         BpL, bias4, cbuf, hb[(t + 1) & 1]);
  }
  unsigned short* hfin = hb[0];   // 256 steps -> even parity -> h0

  // Heads
  k_gemm_bias<<<128, 256, 0, stream>>>(hfin, Bpc1, bc1, u_c, nullptr, 1024, 1024);
  k_gemm_bias<<<128, 256, 0, stream>>>(u_c,  Bpc2, bc2, nullptr, v_c, 1024, 1024);
  k_gemm_bias<<<128, 256, 0, stream>>>(hfin, Bpp1, bp1, u_p, nullptr, 1024, 1024);
  k_gemm_bias<<<128, 256, 0, stream>>>(u_p,  Bpp2, bp2, nullptr, v_p, 1024, 1024);

  int waves = BB * 17;
  k_head_final<<<(waves * 32 + 255) / 256, 256, 0, stream>>>(v_c, wc3, bc3,
                                                             v_p, wp3, bp3, out);
}
